// _AFM_34643206210189
// MI455X (gfx1250) — matrix operand order repaired and checked by reading, not compiled, not run on hardware
//
#include <hip/hip_runtime.h>

#define NFIELD 39
#define EMBD   64
#define ATTND  32
#define NPAIR  741          // 39*38/2
#define NPAD   768
#define NTILE  47           // ceil(741/16)
#define BLOCK  256          // 8 waves (wave32)
#define NWAVE  (BLOCK / 32)
#define ZROW   NFIELD       // zeroed e_s row for padding pairs

typedef __attribute__((ext_vector_type(16))) _Float16 v16h;
typedef __attribute__((ext_vector_type(8)))  float    v8f;

union HFrag { v16h v; _Float16 h[16]; };

__device__ __forceinline__ v8f wmma16x16x32(v16h a, v16h b, v8f c) {
  // D = A(16x32 f16) * B(32x16 f16) + C(16x16 f32)
  return __builtin_amdgcn_wmma_f32_16x16x32_f16(false, a, false, b,
                                                (short)0, c, false, false);
}

__global__ __launch_bounds__(BLOCK) void afm_fused_kernel(
    const int*   __restrict__ fidx,   // (B, 39)
    const float* __restrict__ fval,   // (B, 39)
    const float* __restrict__ w_lin,  // (1M, 1)
    const float* __restrict__ b_lin,  // (1,)
    const float* __restrict__ emb,    // (1M, 64)
    const float* __restrict__ W1,     // (64, 32) row-major
    const float* __restrict__ b1,     // (32,)
    const float* __restrict__ W2,     // (32, 1)
    const float* __restrict__ b2,     // (1,)
    const float* __restrict__ Wfc,    // (64, 1)
    const float* __restrict__ bfc,    // (1,)
    float*       __restrict__ out)    // (B, 1)
{
  __shared__ __align__(16) float e_s[(NFIELD + 1) * EMBD]; // +1 zero row
  __shared__ float s_sc[NPAD];                             // logits -> scores
  __shared__ unsigned char prow[NPAD], pcol[NPAD];
  __shared__ float red[BLOCK];
  __shared__ float attnp[4][EMBD];
  __shared__ float linred[NFIELD + 1];

  const int tid  = threadIdx.x;
  const int b    = blockIdx.x;
  const int lane = tid & 31;
  const int wave = tid >> 5;
  const int hi   = (lane >= 16) ? 1 : 0;          // which 16-lane half
  const int l15  = lane & 15;

  // ---------------- Phase 0: tables, linear term, embedding gather --------
  for (int q = tid; q < NPAD; q += BLOCK) {
    s_sc[q] = -__builtin_inff();                  // padded logits = -inf
    if (q < NPAIR) {
      int rr = 0, rem = q, n = NFIELD - 1;
      while (rem >= n) { rem -= n; --n; ++rr; }   // upper-triangular (k=1)
      prow[q] = (unsigned char)rr;
      pcol[q] = (unsigned char)(rr + 1 + rem);
    } else {
      prow[q] = (unsigned char)ZROW;              // padding -> zero row
      pcol[q] = (unsigned char)ZROW;
    }
  }
  if (tid < NFIELD) {
    int ix = fidx[b * NFIELD + tid];
    linred[tid] = w_lin[ix] * fval[b * NFIELD + tid];
  }
  if (tid < EMBD) e_s[ZROW * EMBD + tid] = 0.0f;  // zero row
  for (int i = tid; i < NFIELD * EMBD; i += BLOCK) {
    int f = i >> 6, d = i & 63;
    size_t ix = (size_t)fidx[b * NFIELD + f];
    e_s[i] = emb[ix * (size_t)EMBD + d] * fval[b * NFIELD + f];
  }
  __syncthreads();

  // ---------------- Phase 1: transposed GEMM: h^T = W1^T @ p^T -------------
  // A fragments hold W1^T (M = attn dim, K = emb dim), preloaded per wave.
  // A layout (16-bit 16x32): lane row = l15, halves h -> K offset
  //   h + 16*hi, fragment (mt, ks) adds m = mt*16, k = ks*32.
  v16h Af[2][2];
  #pragma unroll
  for (int mt = 0; mt < 2; ++mt) {
    #pragma unroll
    for (int ks = 0; ks < 2; ++ks) {
      HFrag af;
      const int m = mt * 16 + l15;
      #pragma unroll
      for (int h = 0; h < 16; ++h) {
        const int kd = ks * 32 + h + hi * 16;
        af.h[h] = (_Float16)W1[kd * ATTND + m];   // W1^T[m][kd]
      }
      Af[mt][ks] = af.v;
    }
  }
  // Per-lane slices of b1 / W2: acc0 covers rows 8*hi..8*hi+7 (mt=0),
  // acc1 covers rows 16+8*hi.. (mt=1).  (32B-aligned float4 loads)
  const float4* b1v4 = (const float4*)b1;
  const float4* w2v4 = (const float4*)W2;
  const float4 b1a0 = b1v4[hi * 2 + 0], b1a1 = b1v4[hi * 2 + 1];
  const float4 b1b0 = b1v4[4 + hi * 2 + 0], b1b1 = b1v4[4 + hi * 2 + 1];
  const float4 w2a0 = w2v4[hi * 2 + 0], w2a1 = w2v4[hi * 2 + 1];
  const float4 w2b0 = w2v4[4 + hi * 2 + 0], w2b1 = w2v4[4 + hi * 2 + 1];
  const float b2v = b2[0];

  for (int tile = wave; tile < NTILE; tile += NWAVE) {
    // B fragment = p^T tile: lane = pair column l15 (padding -> zero row).
    const int q = tile * 16 + l15;                 // always < NPAD
    const int r = prow[q];
    const int c = pcol[q];
    const float4* er4 = (const float4*)&e_s[r * EMBD];
    const float4* ec4 = (const float4*)&e_s[c * EMBD];

    v8f acc0, acc1;                                // C init = b1 rows
    acc0[0]=b1a0.x; acc0[1]=b1a0.y; acc0[2]=b1a0.z; acc0[3]=b1a0.w;
    acc0[4]=b1a1.x; acc0[5]=b1a1.y; acc0[6]=b1a1.z; acc0[7]=b1a1.w;
    acc1[0]=b1b0.x; acc1[1]=b1b0.y; acc1[2]=b1b0.z; acc1[3]=b1b0.w;
    acc1[4]=b1b1.x; acc1[5]=b1b1.y; acc1[6]=b1b1.z; acc1[7]=b1b1.w;

    #pragma unroll
    for (int ks = 0; ks < 2; ++ks) {
      // B layout (16-bit 32x16): lane = N, halves 0..15 -> K = kb + j,
      // kb = ks*32 + 16*hi. 16 contiguous floats -> 4x ds_load_b128 per row.
      HFrag Bv;
      const int kb4 = (ks * 32 + hi * 16) >> 2;    // float4 index
      #pragma unroll
      for (int j4 = 0; j4 < 4; ++j4) {
        const float4 ra = er4[kb4 + j4];
        const float4 ca = ec4[kb4 + j4];
        Bv.h[j4 * 4 + 0] = (_Float16)(ra.x * ca.x);
        Bv.h[j4 * 4 + 1] = (_Float16)(ra.y * ca.y);
        Bv.h[j4 * 4 + 2] = (_Float16)(ra.z * ca.z);
        Bv.h[j4 * 4 + 3] = (_Float16)(ra.w * ca.w);
      }
      acc0 = wmma16x16x32(Af[0][ks], Bv.v, acc0);
      acc1 = wmma16x16x32(Af[1][ks], Bv.v, acc1);
    }

    // ReLU + W2 dot along registers: this lane covers 16 of 32 attn dims
    // for pair l15; partner lane (xor 16) covers the other 16.
    float partial;
    partial  = fmaxf(acc0[0], 0.0f) * w2a0.x;
    partial += fmaxf(acc0[1], 0.0f) * w2a0.y;
    partial += fmaxf(acc0[2], 0.0f) * w2a0.z;
    partial += fmaxf(acc0[3], 0.0f) * w2a0.w;
    partial += fmaxf(acc0[4], 0.0f) * w2a1.x;
    partial += fmaxf(acc0[5], 0.0f) * w2a1.y;
    partial += fmaxf(acc0[6], 0.0f) * w2a1.z;
    partial += fmaxf(acc0[7], 0.0f) * w2a1.w;
    partial += fmaxf(acc1[0], 0.0f) * w2b0.x;
    partial += fmaxf(acc1[1], 0.0f) * w2b0.y;
    partial += fmaxf(acc1[2], 0.0f) * w2b0.z;
    partial += fmaxf(acc1[3], 0.0f) * w2b0.w;
    partial += fmaxf(acc1[4], 0.0f) * w2b1.x;
    partial += fmaxf(acc1[5], 0.0f) * w2b1.y;
    partial += fmaxf(acc1[6], 0.0f) * w2b1.z;
    partial += fmaxf(acc1[7], 0.0f) * w2b1.w;

    const float s_full = partial + __shfl_xor(partial, 16, 32);
    if (hi == 0) {
      s_sc[q] = (q < NPAIR) ? (s_full + b2v) : -__builtin_inff();
    }
  }
  __syncthreads();

  // ---------------- Phase 2: softmax over pairs ----------------------------
  float lm = -__builtin_inff();
  for (int q = tid; q < NPAD; q += BLOCK) lm = fmaxf(lm, s_sc[q]);
  red[tid] = lm;
  __syncthreads();
  #pragma unroll
  for (int s = BLOCK / 2; s > 0; s >>= 1) {
    if (tid < s) red[tid] = fmaxf(red[tid], red[tid + s]);
    __syncthreads();
  }
  const float gmax = red[0];
  __syncthreads();

  float ls = 0.0f;
  for (int q = tid; q < NPAD; q += BLOCK) {
    const float ev = __expf(s_sc[q] - gmax);       // exp(-inf)=0 for padding
    s_sc[q] = ev;
    ls += ev;
  }
  red[tid] = ls;
  __syncthreads();
  #pragma unroll
  for (int s = BLOCK / 2; s > 0; s >>= 1) {
    if (tid < s) red[tid] += red[tid + s];
    __syncthreads();
  }
  const float inv = 1.0f / red[0];
  __syncthreads();
  for (int q = tid; q < NPAD; q += BLOCK) s_sc[q] *= inv;
  __syncthreads();

  // ---------------- Phase 3: attn = sum_q score*p, out linear --------------
  {
    const int d = tid & 63, chunk = tid >> 6;      // 4 pair-chunks x 64 dims
    float acc = 0.0f;
    for (int q = chunk; q < NPAIR; q += 4)
      acc += s_sc[q] * e_s[(int)prow[q] * EMBD + d] * e_s[(int)pcol[q] * EMBD + d];
    attnp[chunk][d] = acc;
  }
  __syncthreads();
  if (tid < EMBD) {
    const float a = attnp[0][tid] + attnp[1][tid] + attnp[2][tid] + attnp[3][tid];
    red[tid] = a * Wfc[tid];
  }
  __syncthreads();
  if (tid == 0) {
    float s = 0.0f;
    #pragma unroll
    for (int d = 0; d < EMBD; ++d) s += red[d];
    float lin = b_lin[0];
    #pragma unroll
    for (int f = 0; f < NFIELD; ++f) lin += linred[f];
    out[b] = s + bfc[0] + lin;
  }
}

extern "C" void kernel_launch(void* const* d_in, const int* in_sizes, int n_in,
                              void* d_out, int out_size, void* d_ws, size_t ws_size,
                              hipStream_t stream) {
  (void)n_in; (void)d_ws; (void)ws_size; (void)out_size;
  const int*   fidx  = (const int*)  d_in[0];
  const float* fval  = (const float*)d_in[1];
  const float* w_lin = (const float*)d_in[2];
  const float* b_lin = (const float*)d_in[3];
  const float* emb   = (const float*)d_in[4];
  const float* W1    = (const float*)d_in[5];
  const float* b1    = (const float*)d_in[6];
  const float* W2    = (const float*)d_in[7];
  const float* b2    = (const float*)d_in[8];
  const float* Wfc   = (const float*)d_in[9];
  const float* bfc   = (const float*)d_in[10];
  float*       out   = (float*)d_out;

  const int B = in_sizes[0] / NFIELD;              // 2048
  afm_fused_kernel<<<B, BLOCK, 0, stream>>>(fidx, fval, w_lin, b_lin, emb,
                                            W1, b1, W2, b2, Wfc, bfc, out);
}
